// SpatialNSA_67130338836485
// MI455X (gfx1250) — compile-verified, matrix operand order checked by reading, not run on hardware
//
#include <hip/hip_runtime.h>
#include <math.h>

// ---------------------------------------------------------------------------
// Types for CDNA5 WMMA (wave32): v_wmma_f32_16x16x32_bf16
// ---------------------------------------------------------------------------
typedef __attribute__((ext_vector_type(16))) __bf16 v16bf;
typedef __attribute__((ext_vector_type(8)))  float  v8f;
typedef __attribute__((ext_vector_type(4)))  unsigned int v4u;

__device__ __forceinline__ unsigned short f2bf(float f) {
    unsigned int u = __float_as_uint(f);
    unsigned int r = u + 0x7FFFu + ((u >> 16) & 1u);   // round-to-nearest-even
    return (unsigned short)(r >> 16);
}

__device__ __forceinline__ v8f wmma_bf16(v16bf a, v16bf b, v8f c) {
    return __builtin_amdgcn_wmma_f32_16x16x32_bf16(
        false, a, false, b, (short)0, c, false, false);
}

// Load a 16x32 bf16 fragment (A layout / transposed-B layout) from an LDS
// tile stored row-major with `stride` (in elements).  Per the CDNA5 ISA:
// lane L holds row m = L&15; K chunks [kb..kb+7] and [kb+16..kb+23] with
// kb = (L>=16 ? 8 : 0).  Two 16-byte loads per lane.
__device__ __forceinline__ v16bf load_frag(const unsigned short* p, int stride) {
    int lane = threadIdx.x & 31;
    int m  = lane & 15;
    int kb = (lane >> 4) << 3;
    union FragU { v16bf v; v4u q[2]; } u;
    u.q[0] = *(const v4u*)(p + m * stride + kb);
    u.q[1] = *(const v4u*)(p + m * stride + kb + 16);
    return u.v;
}

// CDNA5 async global->LDS copy (ISA 15.18.3, tracked by ASYNCcnt).
// VDST = per-lane LDS byte address, VADDR = per-lane 64-bit global address.
__device__ __forceinline__ void async_b128(const unsigned short* lds_dst,
                                           const unsigned short* g_src) {
    unsigned l = (unsigned)(unsigned long long)lds_dst;  // low 32 bits = LDS offset
    asm volatile("global_load_async_to_lds_b128 %0, %1, off"
                 :: "v"(l), "v"(g_src)
                 : "memory");
}
__device__ __forceinline__ void wait_async0() {
    asm volatile("s_wait_asynccnt 0x0" ::: "memory");
}

__device__ __forceinline__ float gelu_exact(float v) {
    return 0.5f * v * (1.0f + erff(v * 0.7071067811865475f));
}

// ---------------------------------------------------------------------------
// Generic fp32-in / fp32-out GEMM with bf16 WMMA compute.
//   C[M,N] = act(Aop[M,K] @ B[K,N] + bias)
//   amode 0: A row-major [M,K]
//   amode 1: A = unfold of qkvbuf (B,4096,768) at channel-offset qkv_off,
//            CBS=4 stride=2 grid (31x31 -> 961 blocks/batch), +pos (optional)
//   amode 2: A = x (B,256,4096) read as xs[b*4096+n][c]  (K must be 256)
// Block: 256 threads (8 waves), tile 128x128, K-step 32.
// ---------------------------------------------------------------------------
__global__ __launch_bounds__(256)
void gemm_bf16_wmma(const float* __restrict__ A, const float* __restrict__ Bw,
                    const float* __restrict__ bias, float* __restrict__ C,
                    int M, int N, int K, int act, int amode,
                    const float* __restrict__ qkvbuf, int qkv_off,
                    const float* __restrict__ pos)
{
    __shared__ alignas(16) unsigned short sA[128 * 32];
    __shared__ alignas(16) unsigned short sB[128 * 32];

    const int t    = threadIdx.x;
    const int lane = t & 31;
    const int wave = t >> 5;
    const int wm   = wave >> 1;          // 0..3 -> 32-row slice
    const int wn   = wave & 1;           // 0..1 -> 64-col slice
    const int rowBase = blockIdx.y * 128;
    const int colBase = blockIdx.x * 128;

    v8f acc[2][4];
    {
        v8f z = {};
        for (int mi = 0; mi < 2; ++mi)
            for (int ni = 0; ni < 4; ++ni) acc[mi][ni] = z;
    }

    for (int k0 = 0; k0 < K; k0 += 32) {
        // ---- stage A tile (128x32) as bf16 ----
        for (int i = 0; i < 16; ++i) {
            int idx = i * 256 + t;
            int r = idx >> 5, c = idx & 31;
            int gr = rowBase + r, gk = k0 + c;
            float v = 0.0f;
            if (gr < M) {
                if (amode == 0) {
                    v = A[(long)gr * K + gk];
                } else if (amode == 2) {
                    // xs[b*4096+n][c] = x[b][c][n]
                    v = A[(((long)(gr >> 12)) * 256 + gk) * 4096L + (gr & 4095)];
                } else {
                    // unfold: row gr -> (b, blk); col gk -> (c, p=kh*4+kw)
                    int b   = gr / 961, blk = gr % 961;
                    int cch = gk >> 4,  p   = gk & 15;
                    int kh  = p >> 2,   kw  = p & 3;
                    int bh  = blk / 31, bw  = blk % 31;
                    int pix = (bh * 2 + kh) * 64 + (bw * 2 + kw);
                    v = qkvbuf[(((long)b * 4096 + pix) * 768) + qkv_off + cch];
                    if (pos) v += pos[gk];
                }
            }
            sA[r * 32 + c] = f2bf(v);
        }
        // ---- stage B tile (32x128) transposed -> sB[n][k] ----
        for (int i = 0; i < 16; ++i) {
            int idx = i * 256 + t;
            int c = idx >> 7, n = idx & 127;
            int gn = colBase + n;
            float v = (gn < N) ? Bw[(long)(k0 + c) * N + gn] : 0.0f;
            sB[n * 32 + c] = f2bf(v);
        }
        if (k0 + 32 < K)                       // -> global_prefetch_b8
            __builtin_prefetch(Bw + (long)(k0 + 32) * N + colBase + (t & 127), 0, 1);
        __syncthreads();

        v16bf af[2], bfv[4];
        af[0] = load_frag(sA + (wm * 32 + 0)  * 32, 32);
        af[1] = load_frag(sA + (wm * 32 + 16) * 32, 32);
        for (int ni = 0; ni < 4; ++ni)
            bfv[ni] = load_frag(sB + (wn * 64 + ni * 16) * 32, 32);

        for (int mi = 0; mi < 2; ++mi)
            for (int ni = 0; ni < 4; ++ni)
                acc[mi][ni] = wmma_bf16(af[mi], bfv[ni], acc[mi][ni]);
        __syncthreads();
    }

    // ---- epilogue: C layout -> lane holds col n, rows mb..mb+7 ----
    const int nIn = lane & 15;
    const int mb  = (lane >> 4) << 3;
    for (int mi = 0; mi < 2; ++mi) {
        for (int ni = 0; ni < 4; ++ni) {
            int gc = colBase + wn * 64 + ni * 16 + nIn;
            if (gc >= N) continue;
            float bv = bias ? bias[gc] : 0.0f;
            for (int j = 0; j < 8; ++j) {
                int gr = rowBase + wm * 32 + mi * 16 + mb + j;
                if (gr >= M) continue;
                float v = acc[mi][ni][j] + bv;
                if (act == 1) v = gelu_exact(v);
                C[(long)gr * N + gc] = v;
            }
        }
    }
}

// ---------------------------------------------------------------------------
// Flash attention (two-pass softmax) with WMMA and async-LDS K/V staging.
//   Q: (B,4096,qStride) fp32, head offset h*64.
//   Kb:  bf16 (B, nkeysPad, 256), zero-padded keys.
//   Vtb: bf16 (B, 4, 64, nkeysPad), transposed + zero-padded.
//   Out: (B,4096,256) fp32.   Optional imp[b*impStride+key] += sum_attn*scale.
// Block 256 threads = 8 waves; each wave owns 16 query rows; key blocks of 32.
// ---------------------------------------------------------------------------
__global__ __launch_bounds__(256)
void flash_attn_wmma(const float* __restrict__ Q, int qStride, long qBatchStride,
                     const unsigned short* __restrict__ Kb,
                     const unsigned short* __restrict__ Vtb,
                     int nkeys, int nkeysPad, float* __restrict__ Out,
                     float* __restrict__ imp, int impStride, float impScale)
{
    __shared__ alignas(16) unsigned short sQ[128 * 64];
    __shared__ alignas(16) unsigned short sK[32 * 64];
    __shared__ alignas(16) unsigned short sVt[64 * 32];
    __shared__ alignas(16) unsigned short sP[8 * 16 * 32];

    const int t = threadIdx.x, lane = t & 31, wave = t >> 5;
    const int b = blockIdx.y >> 2, h = blockIdx.y & 3;
    const int q0 = blockIdx.x * 128;
    const int nIn = lane & 15, mb = (lane >> 4) << 3;

    // stage pre-scaled Q (scale = hd^-0.5 = 0.125)
    for (int i = 0; i < 32; ++i) {
        int idx = i * 256 + t;
        int r = idx >> 6, d = idx & 63;
        float v = Q[qBatchStride * b + (long)(q0 + r) * qStride + h * 64 + d] * 0.125f;
        sQ[r * 64 + d] = f2bf(v);
    }
    __syncthreads();
    v16bf qa0 = load_frag(sQ + wave * 16 * 64 + 0,  64);
    v16bf qa1 = load_frag(sQ + wave * 16 * 64 + 32, 64);

    float m[8], l[8];
    for (int j = 0; j < 8; ++j) { m[j] = -1e30f; l[j] = 0.0f; }
    const int nkp = ((nkeys + 31) >> 5) << 5;

    // async staging coordinates (one b128 = 8 bf16 per thread per tile)
    const int kkK = t >> 3, d0K = (t & 7) << 3;          // sK: 32 keys x 64 d
    const int ddV = t >> 2, kk0V = (t & 3) << 3;         // sVt: 64 d x 32 keys
    const unsigned short* KbBase  = Kb  + (long)b * nkeysPad * 256 + h * 64;
    const unsigned short* VtbBase = Vtb + (((long)b * 4 + h) * 64) * (long)nkeysPad;

    // ---------------- pass 1: row max / row sum ----------------
    for (int kb0 = 0; kb0 < nkp; kb0 += 32) {
        async_b128(sK + kkK * 64 + d0K,
                   KbBase + (long)(kb0 + kkK) * 256 + d0K);
        wait_async0();
        __syncthreads();
        float sb[2][8];
        for (int ks = 0; ks < 2; ++ks) {
            v16bf k0 = load_frag(sK + ks * 16 * 64 + 0,  64);
            v16bf k1 = load_frag(sK + ks * 16 * 64 + 32, 64);
            v8f z = {};
            v8f sacc = wmma_bf16(qa0, k0, z);
            sacc = wmma_bf16(qa1, k1, sacc);
            int key = kb0 + ks * 16 + nIn;
            bool ok = key < nkeys;
            for (int j = 0; j < 8; ++j) sb[ks][j] = ok ? sacc[j] : -1e30f;
        }
        for (int j = 0; j < 8; ++j) {
            float mx = fmaxf(sb[0][j], sb[1][j]);
            for (int d = 1; d < 16; d <<= 1) mx = fmaxf(mx, __shfl_xor(mx, d, 16));
            float mn = fmaxf(m[j], mx);
            float sum = expf(sb[0][j] - mn) + expf(sb[1][j] - mn);
            for (int d = 1; d < 16; d <<= 1) sum += __shfl_xor(sum, d, 16);
            l[j] = l[j] * expf(m[j] - mn) + sum;
            m[j] = mn;
        }
        __syncthreads();
    }

    // ---------------- pass 2: P = exp(S-m), O += P*V, imp ----------------
    v8f oacc[4];
    { v8f z = {}; for (int i = 0; i < 4; ++i) oacc[i] = z; }
    for (int kb0 = 0; kb0 < nkp; kb0 += 32) {
        async_b128(sK + kkK * 64 + d0K,
                   KbBase + (long)(kb0 + kkK) * 256 + d0K);
        async_b128(sVt + ddV * 32 + kk0V,
                   VtbBase + (long)ddV * nkeysPad + kb0 + kk0V);
        wait_async0();
        __syncthreads();
        for (int ks = 0; ks < 2; ++ks) {
            v16bf k0 = load_frag(sK + ks * 16 * 64 + 0,  64);
            v16bf k1 = load_frag(sK + ks * 16 * 64 + 32, 64);
            v8f z = {};
            v8f sacc = wmma_bf16(qa0, k0, z);
            sacc = wmma_bf16(qa1, k1, sacc);
            int key = kb0 + ks * 16 + nIn;
            bool ok = key < nkeys;
            float isum = 0.0f;
            for (int j = 0; j < 8; ++j) {
                float p = ok ? expf(sacc[j] - m[j]) : 0.0f;
                sP[wave * 512 + (mb + j) * 32 + ks * 16 + nIn] = f2bf(p);
                isum += p / fmaxf(l[j], 1e-20f);
            }
            if (imp && ok) atomicAdd(&imp[b * impStride + key], isum * impScale);
        }
        __syncthreads();
        v16bf pf = load_frag(sP + wave * 512, 32);
        for (int t4 = 0; t4 < 4; ++t4) {
            v16bf vf = load_frag(sVt + t4 * 16 * 32, 32);
            oacc[t4] = wmma_bf16(pf, vf, oacc[t4]);
        }
        __syncthreads();
    }

    for (int t4 = 0; t4 < 4; ++t4)
        for (int j = 0; j < 8; ++j) {
            int r = q0 + wave * 16 + mb + j;
            float invl = 1.0f / fmaxf(l[j], 1e-20f);
            Out[((long)b * 4096 + r) * 256 + h * 64 + t4 * 16 + nIn] = oacc[t4][j] * invl;
        }
}

// ---------------------------------------------------------------------------
// 7x7 window attention with relative-position bias (one window+head per block).
// 128 threads = 4 waves; tokens padded 49 -> 64 (pad tokens fully masked).
// ---------------------------------------------------------------------------
__global__ __launch_bounds__(128)
void window_attn_wmma(const float* __restrict__ qkv, const float* __restrict__ qkv_b,
                      const float* __restrict__ bias_table, float* __restrict__ Out)
{
    __shared__ alignas(16) unsigned short sQ[64 * 64];
    __shared__ alignas(16) unsigned short sK[64 * 64];
    __shared__ alignas(16) unsigned short sVt[64 * 64];
    __shared__ alignas(16) unsigned short sP[4 * 16 * 64];

    const int t = threadIdx.x, lane = t & 31, wave = t >> 5;
    const int wid = blockIdx.x, h = blockIdx.y;
    const int b = wid / 100, wrem = wid % 100;
    const int wy = wrem / 10, wx = wrem % 10;
    const int nIn = lane & 15, mb = (lane >> 4) << 3;

    for (int i = 0; i < 32; ++i) {
        int idx = i * 128 + t;
        int tk = idx >> 6, d = idx & 63;
        int ty = tk / 7, tx = tk % 7;
        int py = wy * 7 + ty, px = wx * 7 + tx;
        bool inwin = tk < 49;
        bool valid = inwin && (py < 64) && (px < 64);
        float qv, kv, vv;
        if (valid) {
            long base = ((long)b * 4096 + py * 64 + px) * 768;
            qv = qkv[base + h * 64 + d];
            kv = qkv[base + 256 + h * 64 + d];
            vv = qkv[base + 512 + h * 64 + d];
        } else if (inwin) {                 // zero-padded pixel -> bias only
            qv = qkv_b[h * 64 + d];
            kv = qkv_b[256 + h * 64 + d];
            vv = qkv_b[512 + h * 64 + d];
        } else { qv = 0.0f; kv = 0.0f; vv = 0.0f; }
        sQ[tk * 64 + d]  = f2bf(qv * 0.125f);
        sK[tk * 64 + d]  = f2bf(kv);
        sVt[d * 64 + tk] = f2bf(vv);
    }
    __syncthreads();

    v16bf qa0 = load_frag(sQ + wave * 16 * 64 + 0,  64);
    v16bf qa1 = load_frag(sQ + wave * 16 * 64 + 32, 64);

    float sb[4][8];
    for (int ns = 0; ns < 4; ++ns) {
        v16bf k0 = load_frag(sK + ns * 16 * 64 + 0,  64);
        v16bf k1 = load_frag(sK + ns * 16 * 64 + 32, 64);
        v8f z = {};
        v8f sacc = wmma_bf16(qa0, k0, z);
        sacc = wmma_bf16(qa1, k1, sacc);
        int kc = ns * 16 + nIn;
        for (int j = 0; j < 8; ++j) {
            int r = wave * 16 + mb + j;
            float s = sacc[j];
            if (r < 49 && kc < 49) {
                int ri = r / 7, rj = r % 7, ci = kc / 7, cj = kc % 7;
                s += bias_table[((ri - ci + 6) * 13 + (rj - cj + 6)) * 4 + h];
            } else {
                s = -1e30f;
            }
            sb[ns][j] = s;
        }
    }
    float invl[8];
    for (int j = 0; j < 8; ++j) {
        float mx = fmaxf(fmaxf(sb[0][j], sb[1][j]), fmaxf(sb[2][j], sb[3][j]));
        for (int d = 1; d < 16; d <<= 1) mx = fmaxf(mx, __shfl_xor(mx, d, 16));
        float sum = 0.0f;
        for (int ns = 0; ns < 4; ++ns) {
            float p = expf(sb[ns][j] - mx);
            sb[ns][j] = p;
            sum += p;
        }
        for (int d = 1; d < 16; d <<= 1) sum += __shfl_xor(sum, d, 16);
        invl[j] = 1.0f / fmaxf(sum, 1e-20f);
    }
    for (int ns = 0; ns < 4; ++ns)
        for (int j = 0; j < 8; ++j)
            sP[wave * 1024 + (mb + j) * 64 + ns * 16 + nIn] = f2bf(sb[ns][j]);
    __syncthreads();

    v16bf p0 = load_frag(sP + wave * 1024 + 0,  64);
    v16bf p1 = load_frag(sP + wave * 1024 + 32, 64);
    for (int t4 = 0; t4 < 4; ++t4) {
        v8f z = {};
        v8f acc = wmma_bf16(p0, load_frag(sVt + t4 * 16 * 64 + 0,  64), z);
        acc = wmma_bf16(p1, load_frag(sVt + t4 * 16 * 64 + 32, 64), acc);
        for (int j = 0; j < 8; ++j) {
            int r = wave * 16 + mb + j;
            if (r < 49) {
                int py = wy * 7 + r / 7, px = wx * 7 + r % 7;
                if (py < 64 && px < 64)
                    Out[((long)b * 4096 + py * 64 + px) * 256 + h * 64 + t4 * 16 + nIn]
                        = acc[j] * invl[j];
            }
        }
    }
}

// ---------------------------------------------------------------------------
// Helpers: zero, bf16 K/V packing, top-16 selection, gather, gate+combine
// ---------------------------------------------------------------------------
__global__ void zero_f32(float* p, int n) {
    int i = blockIdx.x * 256 + threadIdx.x;
    if (i < n) p[i] = 0.0f;
}

// K,V fp32 (B,nkeys,256) -> Kb bf16 (B,nkeysPad,256) and Vtb bf16 (B,4,64,nkeysPad)
__global__ __launch_bounds__(256)
void pack_kv_bf16(const float* __restrict__ K, const float* __restrict__ V,
                  unsigned short* __restrict__ Kb, unsigned short* __restrict__ Vtb,
                  int nkeys, int nkeysPad)
{
    long i = (long)blockIdx.x * 256 + threadIdx.x;     // B*nkeysPad*256
    long tot = 4L * nkeysPad * 256;
    if (i >= tot) return;
    int d = (int)(i & 255);
    long r = i >> 8;
    int key = (int)(r % nkeysPad);
    int b   = (int)(r / nkeysPad);
    float kv = 0.0f, vv = 0.0f;
    if (key < nkeys) {
        long s = ((long)b * nkeys + key) * 256 + d;
        kv = K[s]; vv = V[s];
    }
    Kb[i] = f2bf(kv);
    int h = d >> 6, dd = d & 63;
    Vtb[(((long)b * 4 + h) * 64 + dd) * (long)nkeysPad + key] = f2bf(vv);
}

__global__ __launch_bounds__(256)
void topk16(const float* __restrict__ imp, int* __restrict__ idx, int nb, int impStride)
{
    __shared__ float vals[1024];
    __shared__ float bv[256];
    __shared__ int   bi[256];
    int b = blockIdx.x, t = threadIdx.x;
    for (int i = t; i < 1024; i += 256)
        vals[i] = (i < nb) ? imp[b * impStride + i] : -1e30f;
    __syncthreads();
    for (int r = 0; r < 16; ++r) {
        float best = -1e30f; int besti = 0;
        for (int i = t; i < 1024; i += 256)
            if (vals[i] > best) { best = vals[i]; besti = i; }
        bv[t] = best; bi[t] = besti;
        __syncthreads();
        for (int s = 128; s > 0; s >>= 1) {
            if (t < s && bv[t + s] > bv[t]) { bv[t] = bv[t + s]; bi[t] = bi[t + s]; }
            __syncthreads();
        }
        if (t == 0) { idx[b * 16 + r] = bi[0]; vals[bi[0]] = -1e30f; }
        __syncthreads();
    }
}

// Gather selected 4x4 blocks into bf16 Kb_slc (B,256,256) / Vtb_slc (B,4,64,256).
__global__ __launch_bounds__(256)
void gather_slc(const float* __restrict__ qkv_slc, const int* __restrict__ idx,
                unsigned short* __restrict__ Kb, unsigned short* __restrict__ Vtb)
{
    long i = (long)blockIdx.x * 256 + threadIdx.x;   // 4*256*256 elements
    if (i >= 4L * 256 * 256) return;
    int d = (int)(i & 255);
    int row = (int)((i >> 8) & 255);
    int b = (int)(i >> 16);
    int sel = row >> 4, s = row & 15;
    int blk = idx[b * 16 + sel];
    if (blk > 255) blk = 255;                        // jnp.clip(idx, 0, nb2-1)
    int flat = s * 256 + d;
    int c = flat >> 4, p = flat & 15;
    int kh = p >> 2, kw = p & 3;
    int bh = blk >> 4, bw = blk & 15;
    int pix = (bh * 4 + kh) * 64 + (bw * 4 + kw);
    long base = ((long)b * 4096 + pix) * 768;
    Kb[i] = f2bf(qkv_slc[base + 256 + c]);
    int h = d >> 6, dd = d & 63;
    Vtb[(((long)b * 4 + h) * 64 + dd) * 256L + row] = f2bf(qkv_slc[base + 512 + c]);
}

__global__ __launch_bounds__(256)
void gate_combine(const float* __restrict__ g1, const float* __restrict__ w2,
                  const float* __restrict__ b2,
                  const float* __restrict__ oc, const float* __restrict__ os,
                  const float* __restrict__ ow, float* __restrict__ out)
{
    int i = blockIdx.x * 256 + threadIdx.x;          // (b,n) flat, 16384
    if (i >= 4 * 4096) return;
    float a0 = b2[0], a1 = b2[1], a2 = b2[2];
    const float* gr = g1 + (long)i * 64;
    for (int k = 0; k < 64; ++k) {
        float hv = gr[k];
        a0 += hv * w2[k * 3 + 0];
        a1 += hv * w2[k * 3 + 1];
        a2 += hv * w2[k * 3 + 2];
    }
    float s0 = 1.0f / (1.0f + expf(-a0));
    float s1 = 1.0f / (1.0f + expf(-a1));
    float s2 = 1.0f / (1.0f + expf(-a2));
    int b = i >> 12, n = i & 4095;
    long ib = (long)i * 256;
    for (int c = 0; c < 256; ++c) {
        float v = s0 * oc[ib + c] + s1 * os[ib + c] + s2 * ow[ib + c];
        out[((long)b * 256 + c) * 4096L + n] = v;
    }
}

// ---------------------------------------------------------------------------
// Host launcher
// ---------------------------------------------------------------------------
extern "C" void kernel_launch(void* const* d_in, const int* in_sizes, int n_in,
                              void* d_out, int out_size, void* d_ws, size_t ws_size,
                              hipStream_t stream)
{
    (void)in_sizes; (void)n_in; (void)out_size; (void)ws_size;

    const float* x          = (const float*)d_in[0];
    const float* qkv_cmp_w  = (const float*)d_in[1];
    const float* qkv_cmp_b  = (const float*)d_in[2];
    const float* qkv_slc_w  = (const float*)d_in[3];
    const float* qkv_slc_b  = (const float*)d_in[4];
    const float* cmpk_w1    = (const float*)d_in[5];
    const float* cmpk_b1    = (const float*)d_in[6];
    const float* cmpk_w2    = (const float*)d_in[7];
    const float* cmpk_b2    = (const float*)d_in[8];
    const float* cmpv_w1    = (const float*)d_in[9];
    const float* cmpv_b1    = (const float*)d_in[10];
    const float* cmpv_w2    = (const float*)d_in[11];
    const float* cmpv_b2    = (const float*)d_in[12];
    const float* pos_embed  = (const float*)d_in[13];
    const float* win_qkv_w  = (const float*)d_in[14];
    const float* win_qkv_b  = (const float*)d_in[15];
    const float* win_proj_w = (const float*)d_in[16];
    const float* win_proj_b = (const float*)d_in[17];
    const float* win_bias   = (const float*)d_in[18];
    const float* proj_cmp_w = (const float*)d_in[19];
    const float* proj_cmp_b = (const float*)d_in[20];
    const float* proj_slc_w = (const float*)d_in[21];
    const float* proj_slc_b = (const float*)d_in[22];
    const float* gate_w1    = (const float*)d_in[23];
    const float* gate_b1    = (const float*)d_in[24];
    const float* gate_w2    = (const float*)d_in[25];
    const float* gate_b2    = (const float*)d_in[26];

    float* ws = (float*)d_ws;
    long o = 0;
    float* qkv_cmp  = ws + o; o += 16384L * 768;
    float* qkv_slc  = ws + o; o += 16384L * 768;
    float* win_qkv  = ws + o; o += 16384L * 768;
    float* hid_k    = ws + o; o += 3844L * 512;
    float* hid_v    = ws + o; o += 3844L * 512;
    float* k_cmp    = ws + o; o += 3844L * 256;
    float* v_cmp    = ws + o; o += 3844L * 256;
    float* ocmp_pre = ws + o; o += 16384L * 256;
    float* oslc_pre = ws + o; o += 16384L * 256;
    float* owin_pre = ws + o; o += 16384L * 256;
    float* ocmp     = ws + o; o += 16384L * 256;
    float* oslc     = ws + o; o += 16384L * 256;
    float* owin     = ws + o; o += 16384L * 256;
    float* g1       = ws + o; o += 16384L * 64;
    float* imp      = ws + o; o += 4L * 992;
    int*   idx      = (int*)(ws + o); o += 64;
    unsigned short* Kb_cmp  = (unsigned short*)(ws + o); o += 4L * 992 * 256 / 2;
    unsigned short* Vtb_cmp = (unsigned short*)(ws + o); o += 4L * 992 * 256 / 2;
    unsigned short* Kb_slc  = (unsigned short*)(ws + o); o += 4L * 256 * 256 / 2;
    unsigned short* Vtb_slc = (unsigned short*)(ws + o); o += 4L * 256 * 256 / 2;

    dim3 blk(256);
    auto ggrid = [](int M, int N) { return dim3((N + 127) / 128, (M + 127) / 128); };

    // QKV projections (amode 2 reads x as (B,N,C))
    gemm_bf16_wmma<<<ggrid(16384, 768), blk, 0, stream>>>(
        x, qkv_cmp_w, qkv_cmp_b, qkv_cmp, 16384, 768, 256, 0, 2, nullptr, 0, nullptr);
    gemm_bf16_wmma<<<ggrid(16384, 768), blk, 0, stream>>>(
        x, qkv_slc_w, qkv_slc_b, qkv_slc, 16384, 768, 256, 0, 2, nullptr, 0, nullptr);
    gemm_bf16_wmma<<<ggrid(16384, 768), blk, 0, stream>>>(
        x, win_qkv_w, win_qkv_b, win_qkv, 16384, 768, 256, 0, 2, nullptr, 0, nullptr);

    // compress MLPs (unfold fused into A-load; +pos on K path)
    gemm_bf16_wmma<<<ggrid(3844, 512), blk, 0, stream>>>(
        nullptr, cmpk_w1, cmpk_b1, hid_k, 3844, 512, 4096, 1, 1, qkv_cmp, 256, pos_embed);
    gemm_bf16_wmma<<<ggrid(3844, 256), blk, 0, stream>>>(
        hid_k, cmpk_w2, cmpk_b2, k_cmp, 3844, 256, 512, 0, 0, nullptr, 0, nullptr);
    gemm_bf16_wmma<<<ggrid(3844, 512), blk, 0, stream>>>(
        nullptr, cmpv_w1, cmpv_b1, hid_v, 3844, 512, 4096, 1, 1, qkv_cmp, 512, nullptr);
    gemm_bf16_wmma<<<ggrid(3844, 256), blk, 0, stream>>>(
        hid_v, cmpv_w2, cmpv_b2, v_cmp, 3844, 256, 512, 0, 0, nullptr, 0, nullptr);

    // pack K/V to bf16 (+transpose V) for async-LDS staging
    pack_kv_bf16<<<(4 * 992 * 256 + 255) / 256, blk, 0, stream>>>(
        k_cmp, v_cmp, Kb_cmp, Vtb_cmp, 961, 992);

    // compressed attention + importance accumulation
    zero_f32<<<16, blk, 0, stream>>>(imp, 4 * 992);
    flash_attn_wmma<<<dim3(32, 16), blk, 0, stream>>>(
        qkv_cmp, 768, 4096L * 768, Kb_cmp, Vtb_cmp, 961, 992, ocmp_pre,
        imp, 992, 1.0f / 4096.0f);

    // top-k selection + gather (bf16 direct), selected attention
    topk16<<<4, blk, 0, stream>>>(imp, idx, 961, 992);
    gather_slc<<<1024, blk, 0, stream>>>(qkv_slc, idx, Kb_slc, Vtb_slc);
    flash_attn_wmma<<<dim3(32, 16), blk, 0, stream>>>(
        qkv_slc, 768, 4096L * 768, Kb_slc, Vtb_slc, 256, 256, oslc_pre,
        nullptr, 0, 0.0f);

    // window attention
    window_attn_wmma<<<dim3(400, 4), dim3(128), 0, stream>>>(
        win_qkv, win_qkv_b, win_bias, owin_pre);

    // output projections
    gemm_bf16_wmma<<<ggrid(16384, 256), blk, 0, stream>>>(
        ocmp_pre, proj_cmp_w, proj_cmp_b, ocmp, 16384, 256, 256, 0, 0, nullptr, 0, nullptr);
    gemm_bf16_wmma<<<ggrid(16384, 256), blk, 0, stream>>>(
        oslc_pre, proj_slc_w, proj_slc_b, oslc, 16384, 256, 256, 0, 0, nullptr, 0, nullptr);
    gemm_bf16_wmma<<<ggrid(16384, 256), blk, 0, stream>>>(
        owin_pre, win_proj_w, win_proj_b, owin, 16384, 256, 256, 0, 0, nullptr, 0, nullptr);

    // gate MLP layer 1 (GELU), then fused gate2 + sigmoid + combine + NCHW store
    gemm_bf16_wmma<<<ggrid(16384, 64), blk, 0, stream>>>(
        x, gate_w1, gate_b1, g1, 16384, 64, 256, 1, 2, nullptr, 0, nullptr);
    gate_combine<<<64, blk, 0, stream>>>(
        g1, gate_w2, gate_b2, ocmp, oslc, owin, (float*)d_out);
}
